// GAT_34394098106585
// MI455X (gfx1250) — matrix operand order repaired and checked by reading, not compiled, not run on hardware
//
#include <hip/hip_runtime.h>
#include <hip/hip_bf16.h>
#include <math.h>

#define NN   50000
#define EE   1600000
#define FDIM 128      // IN == H*OUT == 128 for both layers
#define NH   4
#define NEG  0.2f
#define LPAD 8        // LDS row padding (halves) to skew banks

typedef __attribute__((ext_vector_type(16))) __bf16 v16bf;
typedef __attribute__((ext_vector_type(8)))  __bf16 v8bf;
typedef __attribute__((ext_vector_type(8)))  float  v8f;

// ---------------------------------------------------------------------------
// GEMM: Hout[N,128] = X[N,128] @ W[128,128], bf16 WMMA with fp32 accumulate.
// One block = 16 rows x 128 cols; 8 waves, one 16x16 tile per wave, K=128.
// W staged TRANSPOSED in LDS so each lane's B fragment is two runs of 8 contiguous
// halves -> two ds_load_b128 (same as the A fragment), no scalar u16 loads.
// ---------------------------------------------------------------------------
__global__ __launch_bounds__(256) void gat_gemm_wmma(const float* __restrict__ X,
                                                     const float* __restrict__ W,
                                                     float* __restrict__ Hout) {
    __shared__ __align__(16) __bf16 Xs[16][FDIM + LPAD];     // ~4.3 KB
    __shared__ __align__(16) __bf16 Wt[FDIM][FDIM + LPAD];   // ~34.8 KB, Wt[n][k] = W[k][n]
    const int tid  = threadIdx.x;
    const int row0 = blockIdx.x * 16;

    for (int i = tid; i < 16 * FDIM; i += 256) {
        int r = i >> 7, k = i & 127;
        Xs[r][k] = (__bf16)X[(row0 + r) * FDIM + k];
    }
    for (int i = tid; i < FDIM * FDIM; i += 256) {
        int k = i >> 7, n = i & 127;            // coalesced global read of W[k][n]
        Wt[n][k] = (__bf16)W[i];                // transposed LDS store
    }
    __syncthreads();

    const int wave = tid >> 5;
    const int lane = tid & 31;
    const int n0   = wave * 16;
    const int row  = lane & 15;           // A-matrix row M = lane % 16
    const int n    = n0 + (lane & 15);    // B/D column

    v8f c = {};
    #pragma unroll
    for (int kk = 0; kk < FDIM; kk += 32) {
        // A fragment (16x32 bf16): lanes 0-15 hold K {0..7,16..23}, lanes 16-31 hold K {8..15,24..31}
        union { v16bf v; v8bf h[2]; } A;
        const int kbA = kk + ((lane < 16) ? 0 : 8);
        A.h[0] = *(const v8bf*)&Xs[row][kbA];
        A.h[1] = *(const v8bf*)&Xs[row][kbA + 16];
        // B fragment (32x16 bf16): lanes 0-15 hold K {0..7,16..23}, lanes 16-31 hold K {8..15,24..31}
        union { v16bf v; v8bf h[2]; } B;
        const int kbB = kk + ((lane < 16) ? 0 : 8);
        B.h[0] = *(const v8bf*)&Wt[n][kbB];
        B.h[1] = *(const v8bf*)&Wt[n][kbB + 16];

        c = __builtin_amdgcn_wmma_f32_16x16x32_bf16(false, A.v, false, B.v,
                                                    (short)0, c, false, false);
    }

    // D layout: VGPR r holds M = r (lanes 0-15) / M = r+8 (lanes 16-31), N = lane%16
    const int moff = (lane < 16) ? 0 : 8;
    #pragma unroll
    for (int r = 0; r < 8; ++r) {
        Hout[(row0 + moff + r) * FDIM + n] = c[r];
    }
}

// ---------------------------------------------------------------------------
// Per-node attention coefficients: a_src/a_dst[n,h] = <h[n,h,:], att_*[h,:]>
// ---------------------------------------------------------------------------
__global__ void gat_attn(const float* __restrict__ Hh, const float* __restrict__ att_s,
                         const float* __restrict__ att_d, float* __restrict__ a_src,
                         float* __restrict__ a_dst) {
    int t = blockIdx.x * blockDim.x + threadIdx.x;   // t = n*NH + h
    if (t >= NN * NH) return;
    int n = t >> 2, h = t & 3;
    const float* hp = Hh + n * FDIM + h * 32;
    const float* as = att_s + h * 32;
    const float* ad = att_d + h * 32;
    float ss = 0.f, sd = 0.f;
    #pragma unroll
    for (int c = 0; c < 32; ++c) { float v = hp[c]; ss += v * as[c]; sd += v * ad[c]; }
    a_src[t] = ss;
    a_dst[t] = sd;
}

__device__ __forceinline__ void atomicMaxF(float* addr, float val) {
    if (val >= 0.f) atomicMax((int*)addr, __float_as_int(val));
    else            atomicMin((unsigned int*)addr, __float_as_uint(val));
}

__device__ __forceinline__ void edge_nodes(const long long* ei, int e, int& s, int& d) {
    if (e < EE) { s = (int)ei[e]; d = (int)ei[EE + e]; }
    else        { s = d = e - EE; }  // self-loops appended
}

// Pass 1: segment max over dst of leakyrelu(a_src[src] + a_dst[dst])
__global__ void gat_edge_max(const long long* __restrict__ ei,
                             const float* __restrict__ a_src, const float* __restrict__ a_dst,
                             float* __restrict__ mmax) {
    int e = blockIdx.x * blockDim.x + threadIdx.x;
    if (e >= EE + NN) return;
    int s, d; edge_nodes(ei, e, s, d);
    float4 as = *(const float4*)(a_src + s * 4);
    float4 ad = *(const float4*)(a_dst + d * 4);
    float v[4] = {as.x + ad.x, as.y + ad.y, as.z + ad.z, as.w + ad.w};
    #pragma unroll
    for (int h = 0; h < 4; ++h) {
        float x = v[h] > 0.f ? v[h] : NEG * v[h];
        atomicMaxF(&mmax[d * 4 + h], x);
    }
}

// Pass 2: segment sum of exp(e - max)
__global__ void gat_edge_sum(const long long* __restrict__ ei,
                             const float* __restrict__ a_src, const float* __restrict__ a_dst,
                             const float* __restrict__ mmax, float* __restrict__ ssum) {
    int e = blockIdx.x * blockDim.x + threadIdx.x;
    if (e >= EE + NN) return;
    int s, d; edge_nodes(ei, e, s, d);
    float4 as = *(const float4*)(a_src + s * 4);
    float4 ad = *(const float4*)(a_dst + d * 4);
    float v[4] = {as.x + ad.x, as.y + ad.y, as.z + ad.z, as.w + ad.w};
    #pragma unroll
    for (int h = 0; h < 4; ++h) {
        float x = v[h] > 0.f ? v[h] : NEG * v[h];
        atomicAdd(&ssum[d * 4 + h], __expf(x - mmax[d * 4 + h]));
    }
}

// Pass 3: out[dst] += h[src] * alpha.  One wave32 per edge: 32 lanes x float4 = 128 ch.
__global__ __launch_bounds__(256) void gat_edge_agg(const long long* __restrict__ ei,
                                                    const float* __restrict__ Hh,
                                                    const float* __restrict__ a_src,
                                                    const float* __restrict__ a_dst,
                                                    const float* __restrict__ mmax,
                                                    const float* __restrict__ ssum,
                                                    float* __restrict__ acc) {
    int e = blockIdx.x * 8 + (threadIdx.x >> 5);
    if (e >= EE + NN) return;
    int lane = threadIdx.x & 31;
    int s, d; edge_nodes(ei, e, s, d);
    int j = lane * 4;            // channel base; all 4 channels share head j>>5
    int h = j >> 5;
    float av = a_src[s * 4 + h] + a_dst[d * 4 + h];
    av = av > 0.f ? av : NEG * av;
    float alpha = __expf(av - mmax[d * 4 + h]) / ssum[d * 4 + h];
    float4 hv = *(const float4*)(Hh + s * FDIM + j);
    float* ap = acc + d * FDIM + j;
    atomicAdd(ap + 0, hv.x * alpha);
    atomicAdd(ap + 1, hv.y * alpha);
    atomicAdd(ap + 2, hv.z * alpha);
    atomicAdd(ap + 3, hv.w * alpha);
}

// x2 = elu(acc + bias)
__global__ void gat_elu_bias(const float* __restrict__ acc, const float* __restrict__ bias,
                             float* __restrict__ x2) {
    int i = blockIdx.x * blockDim.x + threadIdx.x;
    if (i >= NN * FDIM) return;
    float v = acc[i] + bias[i & 127];
    x2[i] = v > 0.f ? v : __expf(v) - 1.f;
}

__global__ void gat_bias_final(float* __restrict__ out, const float* __restrict__ bias) {
    int i = blockIdx.x * blockDim.x + threadIdx.x;
    if (i >= NN * FDIM) return;
    out[i] += bias[i & 127];
}

__global__ void gat_zero(float* __restrict__ p, int n) {
    int i = blockIdx.x * blockDim.x + threadIdx.x;
    if (i < n) p[i] = 0.f;
}

__global__ void gat_init_ms(float* __restrict__ mmax, float* __restrict__ ssum) {
    int i = blockIdx.x * blockDim.x + threadIdx.x;
    if (i >= NN * NH) return;
    mmax[i] = -1e30f;
    ssum[i] = 0.f;
}

extern "C" void kernel_launch(void* const* d_in, const int* in_sizes, int n_in,
                              void* d_out, int out_size, void* d_ws, size_t ws_size,
                              hipStream_t stream) {
    const float*     x   = (const float*)d_in[0];
    const long long* ei  = (const long long*)d_in[1];   // int64 [2, E]
    const float*     W1  = (const float*)d_in[2];
    const float*     as1 = (const float*)d_in[3];
    const float*     ad1 = (const float*)d_in[4];
    const float*     b1  = (const float*)d_in[5];
    const float*     W2  = (const float*)d_in[6];
    const float*     as2 = (const float*)d_in[7];
    const float*     ad2 = (const float*)d_in[8];
    const float*     b2  = (const float*)d_in[9];
    float* out = (float*)d_out;

    // workspace layout (floats): 13.6M floats = 54.4 MB
    float* ws    = (float*)d_ws;
    float* h_buf = ws;                   // N*128
    float* x2    = ws + 6400000;         // N*128
    float* a_src = ws + 12800000;        // N*4
    float* a_dst = ws + 13000000;        // N*4
    float* mmax  = ws + 13200000;        // N*4
    float* ssum  = ws + 13400000;        // N*4

    const int NF   = NN * FDIM;
    const int Etot = EE + NN;
    const int gNF  = (NF + 255) / 256;
    const int gNH  = (NN * NH + 255) / 256;
    const int gE   = (Etot + 255) / 256;
    const int gEw  = (Etot + 7) / 8;     // one wave32 per edge, 8 waves/block
    const int gM   = NN / 16;            // 3125 GEMM row tiles, exact

    // ---------------- layer 1 ----------------
    gat_zero<<<gNF, 256, 0, stream>>>(out, NF);
    gat_init_ms<<<gNH, 256, 0, stream>>>(mmax, ssum);
    gat_gemm_wmma<<<gM, 256, 0, stream>>>(x, W1, h_buf);
    gat_attn<<<gNH, 256, 0, stream>>>(h_buf, as1, ad1, a_src, a_dst);
    gat_edge_max<<<gE, 256, 0, stream>>>(ei, a_src, a_dst, mmax);
    gat_edge_sum<<<gE, 256, 0, stream>>>(ei, a_src, a_dst, mmax, ssum);
    gat_edge_agg<<<gEw, 256, 0, stream>>>(ei, h_buf, a_src, a_dst, mmax, ssum, out);
    gat_elu_bias<<<gNF, 256, 0, stream>>>(out, b1, x2);

    // ---------------- layer 2 ----------------
    gat_zero<<<gNF, 256, 0, stream>>>(out, NF);
    gat_init_ms<<<gNH, 256, 0, stream>>>(mmax, ssum);
    gat_gemm_wmma<<<gM, 256, 0, stream>>>(x2, W2, h_buf);
    gat_attn<<<gNH, 256, 0, stream>>>(h_buf, as2, ad2, a_src, a_dst);
    gat_edge_max<<<gE, 256, 0, stream>>>(ei, a_src, a_dst, mmax);
    gat_edge_sum<<<gE, 256, 0, stream>>>(ei, a_src, a_dst, mmax, ssum);
    gat_edge_agg<<<gEw, 256, 0, stream>>>(ei, h_buf, a_src, a_dst, mmax, ssum, out);
    gat_bias_final<<<gNF, 256, 0, stream>>>(out, b2);
}
